// ComplexExtraction_38963943309890
// MI455X (gfx1250) — compile-verified, matrix operand order checked by reading, not run on hardware
//
// MI455X (gfx1250) implementation of the complex transformer reference.
// All large GEMMs run on v_wmma_f32_16x16x32_bf16 via Karatsuba complex GEMM
// (3 WMMAs per complex tile; bf16 WMMA has no A/B negation per CDNA5 ISA).
// v3: B-matrix pointers as named scalars (not an array) so all fragment loads
// lower to global_load_b128 instead of flat_load_b128 (FLAT burns DScnt and
// the LDS pipe per CDNA5 ISA ch.10; v2 asm showed s_wait_loadcnt_dscnt).
#include <hip/hip_runtime.h>
#include <hip/hip_bf16.h>

#define TLEN 501
#define TP   512
#define NB   4
#define EMB  512
#define NCH  16
#define FDIM 8192
#define NH   8
#define DH   64
#define MPAD (NB*TP)   // 2048 padded rows

typedef __attribute__((ext_vector_type(16))) __bf16 v16bf;
typedef __attribute__((ext_vector_type(8)))  __bf16 v8bf;
typedef __attribute__((ext_vector_type(8)))  float  v8f;

// ---------------- WMMA fragment helpers ----------------------------------
// 16-bit A/B fragment layout (CDNA5 ISA 7.12.2): lane = row (mod 16),
// lanes 0-15 hold K = [k0, k0+8) and [k0+16, k0+24); lanes 16-31 hold +8.
// Pointer already includes (row * ld + lane-K-select); just load two chunks.
__device__ __forceinline__ v16bf ldfrag(const __bf16* __restrict__ p) {
  v8bf lo = *(const v8bf*)(p);
  v8bf hi = *(const v8bf*)(p + 16);
  return __builtin_shufflevector(lo, hi, 0,1,2,3,4,5,6,7,8,9,10,11,12,13,14,15);
}

__device__ __forceinline__ v16bf addfrag(v16bf a, v16bf b) {
  v16bf r;
#pragma unroll
  for (int i = 0; i < 16; ++i) r[i] = (__bf16)((float)a[i] + (float)b[i]);
  return r;
}

__device__ __forceinline__ v8f wmma_bf(v16bf a, v16bf b, v8f c) {
  return __builtin_amdgcn_wmma_f32_16x16x32_bf16(false, a, false, b,
                                                 (short)0, c, false, false);
}

// ---------------- generic complex bf16 GEMM ------------------------------
// C[m,n] = alpha * sum_k A[m,k]*W[n,k] (+bias[n]) (+Madd[m,n]) (CReLU opt.)
// A,W as split re/im bf16 planes. z-batched: z -> (zb = z/zH, zh = z%zH),
// per-plane element offsets zb*OffB + zh*OffH.
// CONTRACT: ntiles % 4 == 0 and K % 32 == 0 (all launches below satisfy it),
// so the WMMA main loop is completely branch-free (EXEC stays all-ones).
struct CGArgs {
  const __bf16 *are, *aim; int lda;
  const __bf16 *wre, *wim; int ldw;
  const float  *bias;              // cfloat[N] or null
  float        *c;   int ldc;      // cfloat, row-major
  const float  *madd; int ldadd;   // optional additive mask (cfloat)
  int   ntiles, K;                 // N tiles (16 cols each); K mult of 32
  float alpha;
  int   relu;
  int   zH;                        // z = zb*zH + zh
  long  aOffB, aOffH, wOffB, wOffH, cOffB, cOffH, mOffB, mOffH;
  int   tvalid;                    // row m valid iff (m & (TP-1)) < tvalid
  int   store_mode;                // 0 = normal, 1 = permuted final store
};

__global__ __launch_bounds__(128) void cgemm_kernel(CGArgs a) {
  const int lane = threadIdx.x & 31;
  const int wave = threadIdx.x >> 5;
  const int z  = blockIdx.z;
  const int zb = z / a.zH;
  const int zh = z % a.zH;

  const __bf16* Are = a.are + (long)zb * a.aOffB + (long)zh * a.aOffH;
  const __bf16* Aim = a.aim + (long)zb * a.aOffB + (long)zh * a.aOffH;
  const __bf16* Wre = a.wre + (long)zb * a.wOffB + (long)zh * a.wOffH;
  const __bf16* Wim = a.wim + (long)zb * a.wOffB + (long)zh * a.wOffH;
  float*        Cp  = a.c   + 2 * ((long)zb * a.cOffB + (long)zh * a.cOffH);
  const float*  Mad = a.madd ? a.madd + 2 * ((long)zb * a.mOffB + (long)zh * a.mOffH)
                             : (const float*)0;

  const int row0 = blockIdx.x * 16;
  const int nt0  = (blockIdx.y * 4 + wave) * 4;   // first of 4 n-tiles
  if (nt0 >= a.ntiles) return;                    // uniform early-out only

  // per-lane base pointers (advanced by 32 elements per K-step)
  const int rsel = lane & 15;
  const int ksel = (lane >> 4) << 3;
  const __bf16* pAre = Are + (long)(row0 + rsel) * a.lda + ksel;
  const __bf16* pAim = Aim + (long)(row0 + rsel) * a.lda + ksel;
  const long w0 = (long)(nt0 * 16 +  0 + rsel) * a.ldw + ksel;
  const long w1 = (long)(nt0 * 16 + 16 + rsel) * a.ldw + ksel;
  const long w2 = (long)(nt0 * 16 + 32 + rsel) * a.ldw + ksel;
  const long w3 = (long)(nt0 * 16 + 48 + rsel) * a.ldw + ksel;
  const __bf16* pW0r = Wre + w0;  const __bf16* pW0i = Wim + w0;
  const __bf16* pW1r = Wre + w1;  const __bf16* pW1i = Wim + w1;
  const __bf16* pW2r = Wre + w2;  const __bf16* pW2i = Wim + w2;
  const __bf16* pW3r = Wre + w3;  const __bf16* pW3i = Wim + w3;

  v8f p1[4] = {}, p2[4] = {}, p3[4] = {};         // Karatsuba accumulators

  for (int k0 = 0; k0 < a.K; k0 += 32) {
    v16bf ar = ldfrag(pAre);  pAre += 32;
    v16bf ai = ldfrag(pAim);  pAim += 32;
    v16bf as = addfrag(ar, ai);

    { v16bf br = ldfrag(pW0r);  pW0r += 32;
      v16bf bi = ldfrag(pW0i);  pW0i += 32;
      v16bf bs = addfrag(br, bi);
      p1[0] = wmma_bf(ar, br, p1[0]);
      p2[0] = wmma_bf(ai, bi, p2[0]);
      p3[0] = wmma_bf(as, bs, p3[0]); }
    { v16bf br = ldfrag(pW1r);  pW1r += 32;
      v16bf bi = ldfrag(pW1i);  pW1i += 32;
      v16bf bs = addfrag(br, bi);
      p1[1] = wmma_bf(ar, br, p1[1]);
      p2[1] = wmma_bf(ai, bi, p2[1]);
      p3[1] = wmma_bf(as, bs, p3[1]); }
    { v16bf br = ldfrag(pW2r);  pW2r += 32;
      v16bf bi = ldfrag(pW2i);  pW2i += 32;
      v16bf bs = addfrag(br, bi);
      p1[2] = wmma_bf(ar, br, p1[2]);
      p2[2] = wmma_bf(ai, bi, p2[2]);
      p3[2] = wmma_bf(as, bs, p3[2]); }
    { v16bf br = ldfrag(pW3r);  pW3r += 32;
      v16bf bi = ldfrag(pW3i);  pW3i += 32;
      v16bf bs = addfrag(br, bi);
      p1[3] = wmma_bf(ar, br, p1[3]);
      p2[3] = wmma_bf(ai, bi, p2[3]);
      p3[3] = wmma_bf(as, bs, p3[3]); }
  }

  // Epilogue.  C/D layout: lane 0-15 col, VGPR j -> row j (+8 for hi lanes).
  const int col = lane & 15;
  const int rb  = (lane >> 4) * 8;
#pragma unroll
  for (int j = 0; j < 4; ++j) {
    int n = (nt0 + j) * 16 + col;
    float bre = 0.f, bim = 0.f;
    if (a.bias) { bre = a.bias[2 * n]; bim = a.bias[2 * n + 1]; }
#pragma unroll
    for (int r = 0; r < 8; ++r) {
      int m = row0 + rb + r;
      float re = (p1[j][r] - p2[j][r]) * a.alpha + bre;
      float im = (p3[j][r] - p1[j][r] - p2[j][r]) * a.alpha + bim;
      if (Mad) {
        long mi = 2 * ((long)m * a.ldadd + n);
        re += Mad[mi]; im += Mad[mi + 1];
      }
      if (a.relu) { re = fmaxf(re, 0.f); im = fmaxf(im, 0.f); }
      if ((m & (TP - 1)) >= a.tvalid) { re = 0.f; im = 0.f; }
      if (a.store_mode == 0) {
        long ci = 2 * ((long)m * a.ldc + n);
        Cp[ci] = re; Cp[ci + 1] = im;
      } else {  // final: out[b,e,c,t] from m=b*TP+t, n=e*16+c
        int b = m >> 9, t = m & (TP - 1);
        if (t < TLEN) {
          int e = n >> 4, cc = n & 15;
          long oi = 2 * ((((long)b * EMB + e) * NCH + cc) * TLEN + t);
          Cp[oi] = re; Cp[oi + 1] = im;
        }
      }
    }
  }
}

// ---------------- conversion / packing kernels ---------------------------
__global__ void cvt_planes_kernel(const float* __restrict__ s,
                                  __bf16* __restrict__ re, __bf16* __restrict__ im,
                                  long n) {
  long i = (long)blockIdx.x * blockDim.x + threadIdx.x;
  if (i < n) { re[i] = (__bf16)s[2 * i]; im[i] = (__bf16)s[2 * i + 1]; }
}

__global__ void cvtT_kernel(const float* __restrict__ s,
                            __bf16* __restrict__ re, __bf16* __restrict__ im,
                            int rows, int cols) {
  long i = (long)blockIdx.x * blockDim.x + threadIdx.x;
  if (i >= (long)rows * cols) return;
  int r = (int)(i / cols), c = (int)(i % cols);
  long o = (long)c * rows + r;
  re[o] = (__bf16)s[2 * i]; im[o] = (__bf16)s[2 * i + 1];
}

// emb_mix[b,e,c,t] -> bf16 planes [b*TP+t, c*EMB+e], zero for t>=TLEN
__global__ void pack_kernel(const float* __restrict__ emb,
                            __bf16* __restrict__ re, __bf16* __restrict__ im) {
  long i = (long)blockIdx.x * blockDim.x + threadIdx.x;
  if (i >= (long)MPAD * FDIM) return;
  int f = (int)(i & (FDIM - 1));
  long row = i >> 13;
  int t = (int)(row & (TP - 1)), b = (int)(row >> 9);
  float vr = 0.f, vi = 0.f;
  if (t < TLEN) {
    int c = f >> 9, e = f & (EMB - 1);
    long si = 2 * ((((long)b * EMB + e) * NCH + c) * TLEN + t);
    vr = emb[si]; vi = emb[si + 1];
  }
  re[i] = (__bf16)vr; im[i] = (__bf16)vi;
}

// bn = x * h (broadcast h over T)
__global__ void bmul_kernel(const float* __restrict__ x, const float* __restrict__ h,
                            float* __restrict__ o) {
  long i = (long)blockIdx.x * blockDim.x + threadIdx.x;
  if (i >= (long)MPAD * EMB) return;
  int n = (int)(i & (EMB - 1));
  int b = (int)((i >> 9) >> 9);
  float xr = x[2 * i], xi = x[2 * i + 1];
  long hi = 2 * ((long)b * EMB + n);
  float hr = h[hi], hh = h[hi + 1];
  o[2 * i]     = xr * hr - xi * hh;
  o[2 * i + 1] = xr * hh + xi * hr;
}

// ---------------- global layer norm (big path) ---------------------------
__global__ void gln_reduce_kernel(const float* __restrict__ x, float* __restrict__ st) {
  int b = blockIdx.x;
  float s = 0.f, s2 = 0.f;
  for (int idx = threadIdx.x; idx < TLEN * EMB; idx += 256) {
    int t = idx / EMB, n = idx % EMB;
    long i = 2 * (((long)b * TP + t) * EMB + n);
    float re = x[i], im = x[i + 1];
    float m = sqrtf(re * re + im * im);
    s += m; s2 += m * m;
  }
  __shared__ float red[512];
  red[threadIdx.x] = s; red[256 + threadIdx.x] = s2;
  __syncthreads();
  for (int o = 128; o > 0; o >>= 1) {
    if (threadIdx.x < o) {
      red[threadIdx.x] += red[threadIdx.x + o];
      red[256 + threadIdx.x] += red[256 + threadIdx.x + o];
    }
    __syncthreads();
  }
  if (threadIdx.x == 0) {
    float cnt = (float)(TLEN * EMB);
    float mean = red[0] / cnt;
    st[2 * b] = mean;
    st[2 * b + 1] = red[256] / cnt - mean * mean;
  }
}

__global__ void gln_apply_kernel(const float* __restrict__ x, const float* __restrict__ st,
                                 const float* __restrict__ g, const float* __restrict__ bb,
                                 float* __restrict__ o) {
  long i = (long)blockIdx.x * blockDim.x + threadIdx.x;
  if (i >= (long)MPAD * EMB) return;
  long row = i >> 9;
  int t = (int)(row & (TP - 1)), b = (int)(row >> 9);
  if (t >= TLEN) { o[2 * i] = 0.f; o[2 * i + 1] = 0.f; return; }
  float re = x[2 * i], im = x[2 * i + 1];
  float mag = sqrtf(re * re + im * im);
  float nm = (mag - st[2 * b]) * rsqrtf(st[2 * b + 1] + 1e-8f);
  float ur = 1.f, ui = 0.f;
  if (mag > 0.f) { ur = re / mag; ui = im / mag; }
  float nr = nm * ur, ni = nm * ui;
  float gr = g[2 * t], gi = g[2 * t + 1];
  o[2 * i]     = nr * gr - ni * gi + bb[2 * t];
  o[2 * i + 1] = nr * gi + ni * gr + bb[2 * t + 1];
}

// ---------------- HRTF branch (tiny: B=4, T=2, E=256, H=8, D=32) ---------
__global__ void hrtf_kernel(const float* __restrict__ hr,
    const float* __restrict__ qw, const float* __restrict__ qb,
    const float* __restrict__ kw, const float* __restrict__ kb,
    const float* __restrict__ vw, const float* __restrict__ vb,
    const float* __restrict__ ow, const float* __restrict__ ob,
    const float* __restrict__ g2, const float* __restrict__ b2,
    const float* __restrict__ fw, const float* __restrict__ fb,
    float* __restrict__ hout) {
  __shared__ float X[2][256][2], Q[2][256][2], Kk[2][256][2], V[2][256][2], O[2][256][2];
  __shared__ float AW[8][2][2][2];
  __shared__ float red[512];
  int b = blockIdx.x, tid = threadIdx.x;
  for (int j = tid; j < 512; j += 256) {
    int t = j >> 8, n = j & 255;
    long si = 2 * (((long)b * 2 + t) * 256 + n);
    X[t][n][0] = hr[si]; X[t][n][1] = hr[si + 1];
  }
  __syncthreads();
  for (int j = tid; j < 512; j += 256) {                 // q, k, v projections
    int t = j >> 8, n = j & 255;
    float qr = qb[2*n], qi = qb[2*n+1], kr = kb[2*n], ki = kb[2*n+1];
    float vr = vb[2*n], vi = vb[2*n+1];
    for (int m = 0; m < 256; ++m) {
      float xr = X[t][m][0], xi = X[t][m][1];
      int wi2 = 2 * (n * 256 + m);
      float wr = qw[wi2], wm = qw[wi2+1];
      qr += xr*wr - xi*wm; qi += xr*wm + xi*wr;
      wr = kw[wi2]; wm = kw[wi2+1];
      kr += xr*wr - xi*wm; ki += xr*wm + xi*wr;
      wr = vw[wi2]; wm = vw[wi2+1];
      vr += xr*wr - xi*wm; vi += xr*wm + xi*wr;
    }
    Q[t][n][0]=qr; Q[t][n][1]=qi; Kk[t][n][0]=kr; Kk[t][n][1]=ki;
    V[t][n][0]=vr; V[t][n][1]=vi;
  }
  __syncthreads();
  if (tid < 32) {                                        // scores + CReLU
    int h = tid >> 2, t = (tid >> 1) & 1, s = tid & 1;
    float ar = 0.f, ai = 0.f;
    for (int d = 0; d < 32; ++d) {
      float qr = Q[t][h*32+d][0], qi = Q[t][h*32+d][1];
      float kr = Kk[s][h*32+d][0], ki = Kk[s][h*32+d][1];
      ar += qr*kr - qi*ki; ai += qr*ki + qi*kr;
    }
    const float sc = 0.17677669529663687f;               // 1/sqrt(32)
    AW[h][t][s][0] = fmaxf(ar * sc, 0.f);
    AW[h][t][s][1] = fmaxf(ai * sc, 0.f);
  }
  __syncthreads();
  for (int j = tid; j < 512; j += 256) {                 // o = aw @ v
    int t = j >> 8, n = j & 255, h = n >> 5;
    float orr = 0.f, oi = 0.f;
    for (int s = 0; s < 2; ++s) {
      float ar = AW[h][t][s][0], ai = AW[h][t][s][1];
      float vr = V[s][n][0], vi = V[s][n][1];
      orr += ar*vr - ai*vi; oi += ar*vi + ai*vr;
    }
    O[t][n][0] = orr; O[t][n][1] = oi;
  }
  __syncthreads();
  for (int j = tid; j < 512; j += 256) {                 // o-projection -> X
    int t = j >> 8, n = j & 255;
    float rr = ob[2*n], ri = ob[2*n+1];
    for (int m = 0; m < 256; ++m) {
      float xr = O[t][m][0], xi = O[t][m][1];
      int wi2 = 2 * (n * 256 + m);
      float wr = ow[wi2], wm = ow[wi2+1];
      rr += xr*wr - xi*wm; ri += xr*wm + xi*wr;
    }
    X[t][n][0] = rr; X[t][n][1] = ri;
  }
  __syncthreads();
  {                                                      // globln over (2,256)
    float s = 0.f, s2 = 0.f;
    for (int j = tid; j < 512; j += 256) {
      int t = j >> 8, n = j & 255;
      float m = sqrtf(X[t][n][0]*X[t][n][0] + X[t][n][1]*X[t][n][1]);
      s += m; s2 += m * m;
    }
    red[tid] = s; red[256 + tid] = s2;
    __syncthreads();
    for (int o = 128; o > 0; o >>= 1) {
      if (tid < o) { red[tid] += red[tid+o]; red[256+tid] += red[256+tid+o]; }
      __syncthreads();
    }
  }
  float mean = red[0] / 512.f;
  float inv  = rsqrtf(red[256] / 512.f - mean * mean + 1e-8f);
  for (int j = tid; j < 512; j += 256) {
    int t = j >> 8, n = j & 255;
    float xr = X[t][n][0], xi = X[t][n][1];
    float mag = sqrtf(xr*xr + xi*xi);
    float nm = (mag - mean) * inv;
    float ur = 1.f, ui = 0.f;
    if (mag > 0.f) { ur = xr / mag; ui = xi / mag; }
    float nr = nm * ur, ni = nm * ui;
    float gr = g2[2*t], gi = g2[2*t+1];
    Q[t][n][0] = nr*gr - ni*gi + b2[2*t];
    Q[t][n][1] = nr*gi + ni*gr + b2[2*t+1];
  }
  __syncthreads();
  for (int j = tid; j < 512; j += 256) {                 // fc_hrtf (512->512)
    int n = j;
    float rr = fb[2*n], ri = fb[2*n+1];
    for (int m = 0; m < 512; ++m) {
      float xr = Q[m >> 8][m & 255][0], xi = Q[m >> 8][m & 255][1];
      int wi2 = 2 * (n * 512 + m);
      float wr = fw[wi2], wm = fw[wi2+1];
      rr += xr*wr - xi*wm; ri += xr*wm + xi*wr;
    }
    hout[2 * ((long)b * EMB + n)]     = rr;
    hout[2 * ((long)b * EMB + n) + 1] = ri;
  }
}

// ---------------- host orchestration -------------------------------------
extern "C" void kernel_launch(void* const* d_in, const int* in_sizes, int n_in,
                              void* d_out, int out_size, void* d_ws, size_t ws_size,
                              hipStream_t stream) {
  (void)in_sizes; (void)n_in; (void)out_size; (void)ws_size;
  auto F = [&](int i) -> const float* { return (const float*)d_in[i]; };

  // jax pytree (sorted-key) flatten order of setup_inputs()
  const int I_EMB = 0, I_HRTF = 1;
  const int I_AH = 34, I_AM = 42;                         // k.b,k.w,o.b,o.w,q.b,q.w,v.b,v.w
  const int I_FC1_B = 50, I_FC1_W = 51, I_FCH_B = 52, I_FCH_W = 53;
  const int I_GLN_BN = 54, I_GLNH_B = 62, I_GLNH_G = 63;
  const int I_OUT_B = 64, I_OUT_W = 65;

  uint8_t* base = (uint8_t*)d_ws; size_t off = 0;
  auto alloc = [&](size_t bytes) -> void* {
    void* p = base + off; off = (off + bytes + 255) & ~(size_t)255; return p;
  };

  __bf16* p0re = (__bf16*)alloc((size_t)MPAD * FDIM * 2);
  __bf16* p0im = (__bf16*)alloc((size_t)MPAD * FDIM * 2);
  __bf16* fc1wre = (__bf16*)alloc((size_t)EMB * FDIM * 2);
  __bf16* fc1wim = (__bf16*)alloc((size_t)EMB * FDIM * 2);
  __bf16* outwre = (__bf16*)alloc((size_t)FDIM * EMB * 2);
  __bf16* outwim = (__bf16*)alloc((size_t)FDIM * EMB * 2);
  __bf16* awt[5][4][2];                                   // [layer][q,k,v,o][re,im]
  for (int L = 0; L < 5; ++L)
    for (int m = 0; m < 4; ++m)
      for (int p = 0; p < 2; ++p)
        awt[L][m][p] = (__bf16*)alloc((size_t)EMB * EMB * 2);
  float* xbuf  = (float*)alloc((size_t)MPAD * EMB * 8);
  float* qbuf  = (float*)alloc((size_t)MPAD * EMB * 8);
  float* kbuf  = (float*)alloc((size_t)MPAD * EMB * 8);
  float* vbuf  = (float*)alloc((size_t)MPAD * EMB * 8);
  float* obuf  = (float*)alloc((size_t)MPAD * EMB * 8);
  float* atout = (float*)alloc((size_t)MPAD * EMB * 8);
  float* bnbuf = (float*)alloc((size_t)MPAD * EMB * 8);
  __bf16* xp0 = (__bf16*)alloc((size_t)MPAD * EMB * 2);
  __bf16* xp1 = (__bf16*)alloc((size_t)MPAD * EMB * 2);
  __bf16* qp0 = (__bf16*)alloc((size_t)MPAD * EMB * 2);
  __bf16* qp1 = (__bf16*)alloc((size_t)MPAD * EMB * 2);
  __bf16* kp0 = (__bf16*)alloc((size_t)MPAD * EMB * 2);
  __bf16* kp1 = (__bf16*)alloc((size_t)MPAD * EMB * 2);
  __bf16* op0 = (__bf16*)alloc((size_t)MPAD * EMB * 2);
  __bf16* op1 = (__bf16*)alloc((size_t)MPAD * EMB * 2);
  __bf16* vT0 = (__bf16*)alloc((size_t)EMB * MPAD * 2);
  __bf16* vT1 = (__bf16*)alloc((size_t)EMB * MPAD * 2);
  __bf16* awp0 = (__bf16*)alloc((size_t)NB * NH * TP * TP * 2);
  __bf16* awp1 = (__bf16*)alloc((size_t)NB * NH * TP * TP * 2);
  float* awA  = (float*)alloc((size_t)NB * NH * TP * TP * 8);
  float* awB  = (float*)alloc((size_t)NB * NH * TP * TP * 8);
  float* hvec = (float*)alloc((size_t)NB * EMB * 8);
  float* stats = (float*)alloc(64);

  auto cvt = [&](const float* s, __bf16* re, __bf16* im, long n) {
    cvt_planes_kernel<<<dim3((unsigned)((n + 255) / 256)), dim3(256), 0, stream>>>(s, re, im, n);
  };
  auto run = [&](const CGArgs& g, int mtiles, int z) {
    dim3 grid((unsigned)mtiles, (unsigned)((g.ntiles + 15) / 16), (unsigned)z);
    cgemm_kernel<<<grid, dim3(128), 0, stream>>>(g);
  };

  // ---- weight planes (bf16 re/im) ----
  cvt(F(I_FC1_W), fc1wre, fc1wim, (long)EMB * FDIM);
  cvt(F(I_OUT_W), outwre, outwim, (long)FDIM * EMB);
  for (int L = 0; L < 5; ++L) {
    int bi = (L == 0) ? I_AM : (2 + (L - 1) * 8);
    int widx[4] = { bi + 5, bi + 1, bi + 7, bi + 3 };     // q.w, k.w, v.w, o.w
    for (int m = 0; m < 4; ++m)
      cvt(F(widx[m]), awt[L][m][0], awt[L][m][1], (long)EMB * EMB);
  }

  // ---- fc1: pack + GEMM ----
  pack_kernel<<<dim3((unsigned)(((long)MPAD * FDIM + 255) / 256)), dim3(256), 0, stream>>>(
      F(I_EMB), p0re, p0im);
  { CGArgs g{}; g.are = p0re; g.aim = p0im; g.lda = FDIM;
    g.wre = fc1wre; g.wim = fc1wim; g.ldw = FDIM;
    g.bias = F(I_FC1_B); g.c = xbuf; g.ldc = EMB;
    g.ntiles = EMB / 16; g.K = FDIM; g.alpha = 1.f; g.zH = 1; g.tvalid = TLEN;
    run(g, MPAD / 16, 1); }

  // ---- one complex MHA layer on the padded [MPAD, EMB] activations ----
  auto attn = [&](const float* xin, int L, const float* mask, float* awout, float* outb) {
    int bi = (L == 0) ? I_AM : (2 + (L - 1) * 8);
    const float* biasv[3] = { F(bi + 4), F(bi + 0), F(bi + 6) };   // q.b,k.b,v.b
    float* projo[3] = { qbuf, kbuf, vbuf };
    cvt(xin, xp0, xp1, (long)MPAD * EMB);
    for (int m = 0; m < 3; ++m) {
      CGArgs g{}; g.are = xp0; g.aim = xp1; g.lda = EMB;
      g.wre = awt[L][m][0]; g.wim = awt[L][m][1]; g.ldw = EMB;
      g.bias = biasv[m]; g.c = projo[m]; g.ldc = EMB;
      g.ntiles = EMB / 16; g.K = EMB; g.alpha = 1.f; g.zH = 1; g.tvalid = TLEN;
      run(g, MPAD / 16, 1);
    }
    cvt(qbuf, qp0, qp1, (long)MPAD * EMB);
    cvt(kbuf, kp0, kp1, (long)MPAD * EMB);
    cvtT_kernel<<<dim3((unsigned)(((long)MPAD * EMB + 255) / 256)), dim3(256), 0, stream>>>(
        vbuf, vT0, vT1, MPAD, EMB);
    { CGArgs g{};                                        // scores = relu(q k^T/sqrt(D)+mask)
      g.are = qp0; g.aim = qp1; g.lda = EMB; g.aOffB = (long)TP * EMB; g.aOffH = DH;
      g.wre = kp0; g.wim = kp1; g.ldw = EMB; g.wOffB = (long)TP * EMB; g.wOffH = DH;
      g.c = awout; g.ldc = TP; g.cOffB = (long)NH * TP * TP; g.cOffH = (long)TP * TP;
      g.madd = mask; g.ldadd = TP; g.mOffB = (long)NH * TP * TP; g.mOffH = (long)TP * TP;
      g.ntiles = TP / 16; g.K = DH; g.alpha = 0.125f; g.relu = 1; g.zH = NH; g.tvalid = TP;
      run(g, TP / 16, NB * NH); }
    cvt(awout, awp0, awp1, (long)NB * NH * TP * TP);
    { CGArgs g{};                                        // o = aw @ v (v via transpose)
      g.are = awp0; g.aim = awp1; g.lda = TP;
      g.aOffB = (long)NH * TP * TP; g.aOffH = (long)TP * TP;
      g.wre = vT0; g.wim = vT1; g.ldw = MPAD; g.wOffB = (long)TP; g.wOffH = (long)DH * MPAD;
      g.c = obuf; g.ldc = EMB; g.cOffB = (long)TP * EMB; g.cOffH = DH;
      g.ntiles = DH / 16; g.K = TP; g.alpha = 1.f; g.zH = NH; g.tvalid = TP;
      run(g, TP / 16, NB * NH); }
    cvt(obuf, op0, op1, (long)MPAD * EMB);
    { CGArgs g{};                                        // output projection
      g.are = op0; g.aim = op1; g.lda = EMB;
      g.wre = awt[L][3][0]; g.wim = awt[L][3][1]; g.ldw = EMB;
      g.bias = F(bi + 2); g.c = outb; g.ldc = EMB;
      g.ntiles = EMB / 16; g.K = EMB; g.alpha = 1.f; g.zH = 1; g.tvalid = TLEN;
      run(g, MPAD / 16, 1); }
  };

  // ---- attn_mix ----
  attn(xbuf, 0, (const float*)0, awA, atout);

  // ---- HRTF branch ----
  hrtf_kernel<<<dim3(NB), dim3(256), 0, stream>>>(F(I_HRTF),
      F(I_AH + 5), F(I_AH + 4), F(I_AH + 1), F(I_AH + 0),
      F(I_AH + 7), F(I_AH + 6), F(I_AH + 3), F(I_AH + 2),
      F(I_GLNH_G), F(I_GLNH_B), F(I_FCH_W), F(I_FCH_B), hvec);

  // ---- bn = x * h ----
  bmul_kernel<<<dim3((unsigned)(((long)MPAD * EMB + 255) / 256)), dim3(256), 0, stream>>>(
      atout, hvec, bnbuf);

  // ---- 4 masked bn layers + globln ----
  const float* maskp = (const float*)0;
  for (int i = 0; i < 4; ++i) {
    float* awo = (i & 1) ? awB : awA;
    attn(bnbuf, 1 + i, maskp, awo, atout);
    gln_reduce_kernel<<<dim3(NB), dim3(256), 0, stream>>>(atout, stats);
    gln_apply_kernel<<<dim3((unsigned)(((long)MPAD * EMB + 255) / 256)), dim3(256), 0, stream>>>(
        atout, stats, F(I_GLN_BN + 2 * i + 1), F(I_GLN_BN + 2 * i), bnbuf);
    maskp = awo;
  }

  // ---- out_proj with fused permuted store into d_out ----
  cvt(bnbuf, xp0, xp1, (long)MPAD * EMB);
  { CGArgs g{}; g.are = xp0; g.aim = xp1; g.lda = EMB;
    g.wre = outwre; g.wim = outwim; g.ldw = EMB;
    g.bias = F(I_OUT_B); g.c = (float*)d_out; g.ldc = FDIM;
    g.ntiles = FDIM / 16; g.K = EMB; g.alpha = 1.f; g.zH = 1; g.tvalid = TLEN;
    g.store_mode = 1;
    run(g, MPAD / 16, 1); }
}
